// TorchMD_ET_25786983645526
// MI455X (gfx1250) — compile-verified, matrix operand order checked by reading, not run on hardware
//
#include <hip/hip_runtime.h>
#include <hip/hip_bf16.h>

#define NNODES 20000
#define NEDGES 640000
#define HID    128
#define NBINS  16

typedef __attribute__((ext_vector_type(16))) _Float16     v16h;
typedef __attribute__((ext_vector_type(8)))  float        v8f;
typedef __attribute__((ext_vector_type(4)))  unsigned int u32x4;
typedef __attribute__((ext_vector_type(4)))  float        f32x4;

// ---- WMMA fragment load: two contiguous 16B chunks (K..K+7, K+16..K+23) ----
__device__ __forceinline__ v16h load_frag16(const _Float16* p) {
    union { u32x4 u[2]; v16h h; } f;
    f.u[0] = *(const u32x4*)(p);
    f.u[1] = *(const u32x4*)(p + 16);
    return f.h;
}

// ---------------- weight prep: f32 [k][n] -> f16 transposed [n][k] ----------
__global__ __launch_bounds__(256) void prep_weights_kernel(
    const float* __restrict__ W1, const float* __restrict__ W2,
    const float* __restrict__ Wq, const float* __restrict__ Wk,
    _Float16* __restrict__ W1t, _Float16* __restrict__ W2t,
    _Float16* __restrict__ Wqt, _Float16* __restrict__ Wkt) {
    int t = blockIdx.x * 256 + threadIdx.x;
    const int S1 = 128 * 256, S2 = 128 * 128;
    if (t < S1) {
        int n = t / 256, k = t % 256;
        W1t[t] = (_Float16)W1[k * 128 + n];
    } else if (t < S1 + S2) {
        int u = t - S1; int n = u / 128, k = u % 128;
        W2t[u] = (_Float16)W2[k * 128 + n];
    } else if (t < S1 + 2 * S2) {
        int u = t - S1 - S2; int n = u / 128, k = u % 128;
        Wqt[u] = (_Float16)Wq[k * 128 + n];
    } else if (t < S1 + 3 * S2) {
        int u = t - S1 - 2 * S2; int n = u / 128, k = u % 128;
        Wkt[u] = (_Float16)Wk[k * 128 + n];
    }
}

// ---------------- init segment max / denom buffers --------------------------
__global__ __launch_bounds__(256) void init_seg_kernel(float* __restrict__ mbuf,
                                                       float* __restrict__ den) {
    int i = blockIdx.x * 256 + threadIdx.x;
    if (i < NNODES * NBINS) { mbuf[i] = -INFINITY; den[i] = 0.f; }
}

// ---------------- fused edge-MLP + q/k projections (rows 0..N-1 only) -------
// 1 wave per 16-row block; 4 independent waves per workgroup.
__global__ __launch_bounds__(128) void mlp_kernel(
    const float* __restrict__ x, const float* __restrict__ pos,
    const long long* __restrict__ ei,
    const _Float16* __restrict__ W1t, const float* __restrict__ b1,
    const _Float16* __restrict__ W2t, const float* __restrict__ b2,
    const _Float16* __restrict__ Wqt, const float* __restrict__ bq,
    const _Float16* __restrict__ Wkt, const float* __restrict__ bk,
    float* __restrict__ e_buf, float* __restrict__ q_buf,
    float* __restrict__ k_buf, float* __restrict__ ev_buf) {
    __shared__ _Float16 lds_h[4][16][256];  // h = [x_src | x_dst] (f16), later reused for e
    __shared__ _Float16 lds_t[4][16][128];  // silu(h@W1+b1) (f16)

    const int lane = threadIdx.x & 31;
    const int w    = threadIdx.x >> 5;
    const int blk  = blockIdx.x * 4 + w;
    const int row0 = blk * 16;
    if (row0 >= NNODES) return;            // no cross-wave sync anywhere

    // ---- gather stage: 2 lanes per row (src half / dst half) ----
    {
        const int m = lane >> 1, half = lane & 1;
        const int r = row0 + m;
        const int s = (int)ei[r];
        const int d = (int)ei[NEDGES + r];
        const float* xr = x + (size_t)(half ? d : s) * HID;
        _Float16* hd = &lds_h[w][m][half * 128];
        for (int c = 0; c < 128; c += 8) {
            f32x4 a0 = *(const f32x4*)(xr + c);
            f32x4 a1 = *(const f32x4*)(xr + c + 4);
            union { _Float16 h[8]; u32x4 u; } pk;
            pk.h[0] = (_Float16)a0.x; pk.h[1] = (_Float16)a0.y;
            pk.h[2] = (_Float16)a0.z; pk.h[3] = (_Float16)a0.w;
            pk.h[4] = (_Float16)a1.x; pk.h[5] = (_Float16)a1.y;
            pk.h[6] = (_Float16)a1.z; pk.h[7] = (_Float16)a1.w;
            *(u32x4*)(hd + c) = pk.u;
        }
        if (half == 0) {  // unit edge vector per row
            float dx = pos[d * 3 + 0] - pos[s * 3 + 0];
            float dy = pos[d * 3 + 1] - pos[s * 3 + 1];
            float dz = pos[d * 3 + 2] - pos[s * 3 + 2];
            float inv = 1.f / (sqrtf(dx * dx + dy * dy + dz * dz) + 1e-8f);
            ev_buf[r * 3 + 0] = dx * inv;
            ev_buf[r * 3 + 1] = dy * inv;
            ev_buf[r * 3 + 2] = dz * inv;
        }
    }

    const _Float16* hbase = &lds_h[w][0][0];
    const _Float16* tbase = &lds_t[w][0][0];
    const int nlo = lane & 15;
    const int mo  = (lane >> 4) << 3;

    // ---- t = silu(h @ W1 + b1) : K=256 ----
    for (int n0 = 0; n0 < 128; n0 += 16) {
        v8f acc = {};
        const _Float16* wrow = W1t + (size_t)(n0 + nlo) * 256;
        for (int k0 = 0; k0 < 256; k0 += 32) {
            int kb = k0 + mo;  // mo == 8*(lane>>4)
            v16h a = load_frag16(hbase + nlo * 256 + kb);
            v16h b = load_frag16(wrow + kb);
            acc = __builtin_amdgcn_wmma_f32_16x16x32_f16(false, a, false, b,
                                                         (short)0, acc, false, false);
        }
        int n = n0 + nlo; float bias = b1[n];
        for (int i = 0; i < 8; ++i) {
            float v = acc[i] + bias;
            v = v / (1.f + __expf(-v));          // silu
            lds_t[w][mo + i][n] = (_Float16)v;
        }
    }

    // ---- e = t @ W2 + b2 : K=128 ; keep f32 in global, f16 in LDS (reuse h) --
    for (int n0 = 0; n0 < 128; n0 += 16) {
        v8f acc = {};
        const _Float16* wrow = W2t + (size_t)(n0 + nlo) * 128;
        for (int k0 = 0; k0 < 128; k0 += 32) {
            int kb = k0 + mo;
            v16h a = load_frag16(tbase + nlo * 128 + kb);
            v16h b = load_frag16(wrow + kb);
            acc = __builtin_amdgcn_wmma_f32_16x16x32_f16(false, a, false, b,
                                                         (short)0, acc, false, false);
        }
        int n = n0 + nlo; float bias = b2[n];
        for (int i = 0; i < 8; ++i) {
            float v = acc[i] + bias;
            e_buf[(size_t)(row0 + mo + i) * HID + n] = v;
            lds_h[w][mo + i][n] = (_Float16)v;   // e (f16), row stride 256
        }
    }

    // ---- q = e @ Wq + bq ----
    for (int n0 = 0; n0 < 128; n0 += 16) {
        v8f acc = {};
        const _Float16* wrow = Wqt + (size_t)(n0 + nlo) * 128;
        for (int k0 = 0; k0 < 128; k0 += 32) {
            int kb = k0 + mo;
            v16h a = load_frag16(hbase + nlo * 256 + kb);
            v16h b = load_frag16(wrow + kb);
            acc = __builtin_amdgcn_wmma_f32_16x16x32_f16(false, a, false, b,
                                                         (short)0, acc, false, false);
        }
        int n = n0 + nlo; float bias = bq[n];
        for (int i = 0; i < 8; ++i)
            q_buf[(size_t)(row0 + mo + i) * HID + n] = acc[i] + bias;
    }

    // ---- k = e @ Wk + bk ----
    for (int n0 = 0; n0 < 128; n0 += 16) {
        v8f acc = {};
        const _Float16* wrow = Wkt + (size_t)(n0 + nlo) * 128;
        for (int k0 = 0; k0 < 128; k0 += 32) {
            int kb = k0 + mo;
            v16h a = load_frag16(hbase + nlo * 256 + kb);
            v16h b = load_frag16(wrow + kb);
            acc = __builtin_amdgcn_wmma_f32_16x16x32_f16(false, a, false, b,
                                                         (short)0, acc, false, false);
        }
        int n = n0 + nlo; float bias = bk[n];
        for (int i = 0; i < 8; ++i)
            k_buf[(size_t)(row0 + mo + i) * HID + n] = acc[i] + bias;
    }
}

// ---------------- per-edge logits + bin + segment max (CAS float max) -------
__device__ __forceinline__ void atomicMaxF(float* addr, float val) {
    int* ia = (int*)addr;
    int old = *ia;
    while (__int_as_float(old) < val) {
        int assumed = old;
        old = atomicCAS(ia, assumed, __float_as_int(val));
        if (old == assumed) break;
    }
}

__global__ __launch_bounds__(256) void attn_logits_kernel(
    const long long* __restrict__ ei, const float* __restrict__ qb,
    const float* __restrict__ kb, const float* __restrict__ evb,
    float* __restrict__ logit, int* __restrict__ grp, float* __restrict__ mbuf) {
    const int lane = threadIdx.x & 31;
    const int e = blockIdx.x * 8 + (threadIdx.x >> 5);
    if (e >= NEDGES) return;
    const int s = (int)ei[e];
    const int d = (int)ei[NEDGES + e];
    f32x4 qv = *(const f32x4*)(qb + (size_t)d * HID + lane * 4);
    f32x4 kv = *(const f32x4*)(kb + (size_t)s * HID + lane * 4);
    float part = qv.x * kv.x + qv.y * kv.y + qv.z * kv.z + qv.w * kv.w;
    for (int off = 16; off; off >>= 1) part += __shfl_xor(part, off, 32);
    if (lane == 0) {
        float ax = evb[d * 3], ay = evb[d * 3 + 1], az = evb[d * 3 + 2];
        float bx = evb[s * 3], by = evb[s * 3 + 1], bz = evb[s * 3 + 2];
        float cosv = fminf(1.f, fmaxf(-1.f, ax * bx + ay * by + az * bz));
        int bin = (int)ceilf((cosv + 1.f) * 8.f) - 1;   // searchsorted(left)-1
        bin = min(NBINS - 1, max(0, bin));
        int g = d * NBINS + bin;
        float lg = part * 0.08838834764831845f;         // 1/sqrt(128)
        grp[e] = g;
        logit[e] = lg;
        atomicMaxF(mbuf + g, lg);
    }
}

// ---------------- exp + segment denominator ---------------------------------
__global__ __launch_bounds__(256) void attn_exp_kernel(
    const float* __restrict__ logit, const int* __restrict__ grp,
    const float* __restrict__ mbuf, float* __restrict__ exbuf,
    float* __restrict__ den) {
    int e = blockIdx.x * 256 + threadIdx.x;
    if (e >= NEDGES) return;
    int g = grp[e];
    float ex = __expf(logit[e] - mbuf[g]);
    exbuf[e] = ex;
    atomicAdd(den + g, ex);
}

// ---------------- weighted scatter into [N, H*NB] output --------------------
__global__ __launch_bounds__(256) void scatter_kernel(
    const long long* __restrict__ ei, const int* __restrict__ grp,
    const float* __restrict__ exbuf, const float* __restrict__ den,
    const float* __restrict__ ebuf, float* __restrict__ out) {
    const int lane = threadIdx.x & 31;
    const int e = blockIdx.x * 8 + (threadIdx.x >> 5);
    if (e >= NEDGES) return;
    const int s = (int)ei[e];
    const int g = grp[e];
    float attn = exbuf[e] / (den[g] + 1e-16f);
    const int node = g >> 4, bin = g & 15;
    const float* ej = ebuf + (size_t)s * HID;
    float* op = out + (size_t)node * (HID * NBINS) + bin;
    for (int i = 0; i < 4; ++i) {
        int h = lane + 32 * i;
        atomicAdd(op + (size_t)h * NBINS, ej[h] * attn);
    }
}

extern "C" void kernel_launch(void* const* d_in, const int* in_sizes, int n_in,
                              void* d_out, int out_size, void* d_ws, size_t ws_size,
                              hipStream_t stream) {
    const float*     x   = (const float*)d_in[0];
    const float*     pos = (const float*)d_in[1];
    const long long* ei  = (const long long*)d_in[2];
    const float*     W1  = (const float*)d_in[3];
    const float*     b1  = (const float*)d_in[4];
    const float*     W2  = (const float*)d_in[5];
    const float*     b2  = (const float*)d_in[6];
    const float*     Wq  = (const float*)d_in[7];
    const float*     bq  = (const float*)d_in[8];
    const float*     Wk  = (const float*)d_in[9];
    const float*     bk  = (const float*)d_in[10];
    float* out = (float*)d_out;

    // workspace carve (256B aligned)
    char* p = (char*)d_ws;
    auto carve = [&](size_t bytes) -> void* {
        void* r = (void*)p;
        p += (bytes + 255) & ~(size_t)255;
        return r;
    };
    _Float16* W1t    = (_Float16*)carve(128 * 256 * sizeof(_Float16));
    _Float16* W2t    = (_Float16*)carve(128 * 128 * sizeof(_Float16));
    _Float16* Wqt    = (_Float16*)carve(128 * 128 * sizeof(_Float16));
    _Float16* Wkt    = (_Float16*)carve(128 * 128 * sizeof(_Float16));
    float*    e_buf  = (float*)carve((size_t)NNODES * HID * sizeof(float));
    float*    q_buf  = (float*)carve((size_t)NNODES * HID * sizeof(float));
    float*    k_buf  = (float*)carve((size_t)NNODES * HID * sizeof(float));
    float*    ev_buf = (float*)carve((size_t)NNODES * 3 * sizeof(float));
    float*    m_buf  = (float*)carve((size_t)NNODES * NBINS * sizeof(float));
    float*    den    = (float*)carve((size_t)NNODES * NBINS * sizeof(float));
    float*    logit  = (float*)carve((size_t)NEDGES * sizeof(float));
    float*    exbuf  = (float*)carve((size_t)NEDGES * sizeof(float));
    int*      grp    = (int*)carve((size_t)NEDGES * sizeof(int));

    hipMemsetAsync(d_out, 0, (size_t)out_size * sizeof(float), stream);

    prep_weights_kernel<<<(128 * 256 + 3 * 128 * 128 + 255) / 256, 256, 0, stream>>>(
        W1, W2, Wq, Wk, W1t, W2t, Wqt, Wkt);
    init_seg_kernel<<<(NNODES * NBINS + 255) / 256, 256, 0, stream>>>(m_buf, den);

    // 1250 row blocks of 16, 4 waves (blocks) per workgroup
    mlp_kernel<<<(1250 + 3) / 4, 128, 0, stream>>>(
        x, pos, ei, W1t, b1, W2t, b2, Wqt, bq, Wkt, bk,
        e_buf, q_buf, k_buf, ev_buf);

    attn_logits_kernel<<<(NEDGES + 7) / 8, 256, 0, stream>>>(
        ei, q_buf, k_buf, ev_buf, logit, grp, m_buf);
    attn_exp_kernel<<<(NEDGES + 255) / 256, 256, 0, stream>>>(
        logit, grp, m_buf, exbuf, den);
    scatter_kernel<<<(NEDGES + 7) / 8, 256, 0, stream>>>(
        ei, grp, exbuf, den, e_buf, out);
}
